// Self_attention_cnn_72361609003218
// MI455X (gfx1250) — matrix operand order repaired and checked by reading, not compiled, not run on hardware
//
#include <hip/hip_runtime.h>

#define BATCH 16
#define SEQ   2048
#define DMODEL 768

static constexpr float SCALE = 0.03608439182435161f;  // 1/sqrt(768)
static constexpr float LOG2E = 1.4426950408889634f;

typedef __bf16 v16bf __attribute__((ext_vector_type(16)));
typedef float  v8f   __attribute__((ext_vector_type(8)));

// 16-byte int vector type matching the async-copy builtin's parameter type
typedef int v4i_vs __attribute__((__vector_size__(16)));
typedef __attribute__((address_space(1))) v4i_vs* async_gptr;
typedef __attribute__((address_space(3))) v4i_vs* async_lptr;

#if __has_builtin(__builtin_amdgcn_global_load_async_to_lds_b128)
#define HAVE_ASYNC_LDS 1
#else
#define HAVE_ASYNC_LDS 0
#endif

union Frag {
    v16bf        v;
    unsigned int u[8];
};

// native f32 -> bf16 convert (lowers to v_cvt_*_bf16_f32, pairs fuse to cvt_pk)
__device__ __forceinline__ unsigned short f2bf(float x) {
    union { __bf16 h; unsigned short s; } u;
    u.h = (__bf16)x;
    return u.s;
}

__device__ __forceinline__ unsigned int pack_bf(float lo, float hi) {
    union { __bf16 h[2]; unsigned int u; } p;
    p.h[0] = (__bf16)lo;
    p.h[1] = (__bf16)hi;
    return p.u;
}

__device__ __forceinline__ unsigned int wang_hash(unsigned int s) {
    s = (s ^ 61u) ^ (s >> 16);
    s *= 9u;
    s ^= s >> 4;
    s *= 0x27d4eb2du;
    s ^= s >> 15;
    return s;
}

__device__ __forceinline__ void wait_asynccnt0() {
#if __has_builtin(__builtin_amdgcn_s_wait_asynccnt)
    __builtin_amdgcn_s_wait_asynccnt(0);
#else
    asm volatile("s_wait_asynccnt 0" ::: "memory");
#endif
}

__global__ __launch_bounds__(256)
void flash_attn_drop_kernel(const float* __restrict__ x, float* __restrict__ out) {
    // LDS: Q tile bf16 (24KB) + double-buffered K chunk fp32 (64KB) + P tile (4KB) + red (1KB)
    __shared__ unsigned short Qs[16 * DMODEL];
    __shared__ float          Ks[2][128 * 64];
    __shared__ unsigned short Ps[16 * 128];
    __shared__ float redA[8][16];
    __shared__ float redB[8][16];

    const int tid  = threadIdx.x;
    const int w    = tid >> 5;        // wave id 0..7
    const int lane = tid & 31;
    const int half = lane >> 4;       // 0 or 1 (wave32 half)
    const int ln   = lane & 15;

    const int blk = blockIdx.x;
    const int b   = blk >> 7;                 // 128 q-tiles per batch
    const int q0  = (blk & 127) << 4;

    const float* xb = x + (size_t)b * SEQ * DMODEL;

    // ---- load 16x768 Q tile into LDS as bf16 (rows are contiguous in memory) ----
    {
        const float4* qsrc = (const float4*)(xb + (size_t)q0 * DMODEL);
        for (int i = tid; i < (16 * DMODEL) / 4; i += 256) {
            float4 v = qsrc[i];
            int o = i * 4;
            Qs[o + 0] = f2bf(v.x);
            Qs[o + 1] = f2bf(v.y);
            Qs[o + 2] = f2bf(v.z);
            Qs[o + 3] = f2bf(v.w);
        }
    }

    float m_run[8], l_run[8];
    v8f o_acc[6];
#pragma unroll
    for (int r = 0; r < 8; ++r) { m_run[r] = -1e30f; l_run[r] = 0.0f; }
#pragma unroll
    for (int t = 0; t < 6; ++t) { v8f z = {}; o_acc[t] = z; }

    for (int kv = 0; kv < SEQ; kv += 128) {
        // issue K-chunk copies [128 keys x 64 d] of chunk dc into buffer buf
        auto issue_chunk = [&](int dc, int buf) {
            for (int i = tid; i < 2048; i += 256) {
                int krow = i >> 4;
                int d4   = i & 15;
                const float* gsrc = xb + (size_t)(kv + krow) * DMODEL + dc * 64 + d4 * 4;
                float* ldst = &Ks[buf][krow * 64 + d4 * 4];
#if HAVE_ASYNC_LDS
                // CDNA5 async direct global->LDS copy, tracked by ASYNCcnt
                __builtin_amdgcn_global_load_async_to_lds_b128(
                    (async_gptr)gsrc, (async_lptr)ldst, 0, 0);
#else
                *(float4*)ldst = *(const float4*)gsrc;
#endif
            }
        };

        // prefetch next KV block's K rows into cache hierarchy
        if (kv + 128 < SEQ && lane == 0) {
            __builtin_prefetch(xb + (size_t)(kv + 128 + w * 16) * DMODEL, 0, 1);
        }

        // ---- S = Q * K^T, K streamed in 64-wide D chunks, double-buffered async DMA ----
        v8f c = {};
        issue_chunk(0, 0);
        for (int dc = 0; dc < DMODEL / 64; ++dc) {
            const int cur = dc & 1;
#if HAVE_ASYNC_LDS
            wait_asynccnt0();          // our copies for chunk dc have landed
#endif
            __syncthreads();           // everyone's copies for chunk dc visible
            if (dc + 1 < DMODEL / 64) {
                issue_chunk(dc + 1, cur ^ 1);  // overlap next DMA with this chunk's WMMAs
            }
#pragma unroll
            for (int t = 0; t < 2; ++t) {
                Frag a, bm;
#pragma unroll
                for (int j = 0; j < 8; ++j) {
                    // A (16x32 bf16) layout: lane half selects K sub-range
                    int kp = (j < 4) ? (half * 8 + 2 * j) : (16 + half * 8 + 2 * (j - 4));
                    a.u[j] = *(const unsigned int*)&Qs[ln * DMODEL + dc * 64 + t * 32 + kp];
                    // B (32x16 bf16) = K^T: col = key (lane), K dim = d; convert fp32->bf16 here
                    int kk = (j < 4) ? (half * 8 + 2 * j) : (16 + half * 8 + 2 * (j - 4));
                    const float* kptr = &Ks[cur][(w * 16 + ln) * 64 + t * 32 + kk];
                    bm.u[j] = pack_bf(kptr[0], kptr[1]);
                }
                c = __builtin_amdgcn_wmma_f32_16x16x32_bf16(false, a.v, false, bm.v,
                                                            (short)0, c, false, false);
            }
        }

        // ---- online softmax across the 128-key block (8 waves x 16 keys) ----
        float sv[8];
#pragma unroll
        for (int r = 0; r < 8; ++r) sv[r] = c[r] * SCALE;

        float mt[8];
#pragma unroll
        for (int r = 0; r < 8; ++r) {
            float m = sv[r];
            for (int mask = 8; mask >= 1; mask >>= 1) m = fmaxf(m, __shfl_xor(m, mask));
            mt[r] = m;  // row max of this wave's 16-col tile (per lane-half row r / r+8)
        }
        if (lane == 0) {
#pragma unroll
            for (int r = 0; r < 8; ++r) redA[w][r] = mt[r];
        }
        if (lane == 16) {
#pragma unroll
            for (int r = 0; r < 8; ++r) redA[w][8 + r] = mt[r];
        }
        __syncthreads();

        float f[8];
#pragma unroll
        for (int r = 0; r < 8; ++r) {
            float g = -1e30f;
#pragma unroll
            for (int ww = 0; ww < 8; ++ww) g = fmaxf(g, redA[ww][(half << 3) + r]);
            float nm = fmaxf(m_run[r], g);
            f[r] = exp2f((m_run[r] - nm) * LOG2E);
            m_run[r] = nm;
        }

        // p = exp(s - m); dropout folded into P (not into l)
        float ps[8];
#pragma unroll
        for (int r = 0; r < 8; ++r) {
            float p = exp2f((sv[r] - m_run[r]) * LOG2E);
            ps[r] = p;
            int rowg = (half << 3) + r;
            unsigned int idx =
                (unsigned int)(b * SEQ + q0 + rowg) * (unsigned int)SEQ +
                (unsigned int)(kv + w * 16 + ln);
            unsigned int h = wang_hash(idx);
            float pd = ((h & 0xFFFFu) < 32768u) ? p * 2.0f : 0.0f;  // keep, /(1-p)
            Ps[rowg * 128 + w * 16 + ln] = f2bf(pd);
        }

        float st[8];
#pragma unroll
        for (int r = 0; r < 8; ++r) {
            float s = ps[r];
            for (int mask = 8; mask >= 1; mask >>= 1) s += __shfl_xor(s, mask);
            st[r] = s;
        }
        if (lane == 0) {
#pragma unroll
            for (int r = 0; r < 8; ++r) redB[w][r] = st[r];
        }
        if (lane == 16) {
#pragma unroll
            for (int r = 0; r < 8; ++r) redB[w][8 + r] = st[r];
        }
        __syncthreads();

#pragma unroll
        for (int r = 0; r < 8; ++r) {
            float g = 0.0f;
#pragma unroll
            for (int ww = 0; ww < 8; ++ww) g += redB[ww][(half << 3) + r];
            l_run[r] = l_run[r] * f[r] + g;
        }
        // rescale running output
#pragma unroll
        for (int t = 0; t < 6; ++t) {
#pragma unroll
            for (int r = 0; r < 8; ++r) o_acc[t][r] *= f[r];
        }

        // ---- O += P * V, each wave owns a 96-wide D slice; V read from global (L2-hot) ----
        for (int kk = 0; kk < 4; ++kk) {
            Frag ap;
#pragma unroll
            for (int j = 0; j < 8; ++j) {
                int kp = (j < 4) ? (half * 8 + 2 * j) : (16 + half * 8 + 2 * (j - 4));
                ap.u[j] = *(const unsigned int*)&Ps[ln * 128 + kk * 32 + kp];
            }
            const float* vbase = xb + (size_t)(kv + kk * 32) * DMODEL;
            for (int dt = 0; dt < 6; ++dt) {
                int dcol = w * 96 + dt * 16 + ln;
                Frag bv;
#pragma unroll
                for (int j = 0; j < 8; ++j) {
                    int kr = (j < 4) ? (half * 8 + 2 * j) : (16 + half * 8 + 2 * (j - 4));
                    float x0 = vbase[(size_t)(kr) * DMODEL + dcol];
                    float x1 = vbase[(size_t)(kr + 1) * DMODEL + dcol];
                    bv.u[j] = pack_bf(x0, x1);
                }
                o_acc[dt] = __builtin_amdgcn_wmma_f32_16x16x32_bf16(false, ap.v, false, bv.v,
                                                                    (short)0, o_acc[dt],
                                                                    false, false);
            }
        }
    }

    // ---- finalize: divide by softmax denominator and store fp32 ----
    float inv[8];
#pragma unroll
    for (int r = 0; r < 8; ++r) inv[r] = 1.0f / l_run[r];

#pragma unroll
    for (int dt = 0; dt < 6; ++dt) {
        int dcol = w * 96 + dt * 16 + ln;
#pragma unroll
        for (int r = 0; r < 8; ++r) {
            int rowg = q0 + (half << 3) + r;
            out[((size_t)b * SEQ + rowg) * DMODEL + dcol] = o_acc[dt][r] * inv[r];
        }
    }
}

extern "C" void kernel_launch(void* const* d_in, const int* in_sizes, int n_in,
                              void* d_out, int out_size, void* d_ws, size_t ws_size,
                              hipStream_t stream) {
    (void)in_sizes; (void)n_in; (void)out_size; (void)d_ws; (void)ws_size;
    const float* x = (const float*)d_in[0];
    float* out = (float*)d_out;
    dim3 grid(BATCH * (SEQ / 16));   // 2048 workgroups
    dim3 block(256);                 // 8 wave32 waves
    flash_attn_drop_kernel<<<grid, block, 0, stream>>>(x, out);
}
